// final_model_HAN_80668075754166
// MI455X (gfx1250) — matrix operand order repaired and checked by reading, not compiled, not run on hardware
//
#include <hip/hip_runtime.h>
#include <cstdint>

// ---------------------------------------------------------------------------
// Types matching CDNA5 WMMA operand shapes (wave32)
// ---------------------------------------------------------------------------
typedef __attribute__((ext_vector_type(16))) __bf16 v16bf;
typedef __attribute__((ext_vector_type(8)))  __bf16 v8bf;
typedef __attribute__((ext_vector_type(8)))  float  v8f;

#if defined(__HIP_DEVICE_COMPILE__) && __has_builtin(__builtin_amdgcn_global_load_async_to_lds_b128)
#define HAVE_ASYNC 1
#else
#define HAVE_ASYNC 0
#endif

// Problem dims
constexpr int  Bt  = 4;
constexpr int  Nn  = 8192;
constexpr int  Kd  = 128;
constexpr long NK  = (long)Bt * Nn * Kd;   // 4,194,304 elems per z array
constexpr long ADJ = (long)Nn * Nn;        // 67,108,864 adj elems

// Tiling
constexpr int TILE = 128;                  // output tile 128x128 per block
constexpr int LDST = 136;                  // padded LDS row stride (bf16): 272B = 68 dw -> no bank conflicts
constexpr int MATE = TILE * LDST;          // bf16 elems per staged matrix tile
constexpr int BUFE = 2 * MATE;             // A tile + B tile per buffer
constexpr int SMEM_BYTES = 2 * BUFE * 2;   // double buffered, 2B/elem = 139,264 B (<320KB WGP LDS)

__device__ inline unsigned short f2bf(float f) {  // RNE f32 -> bf16
  unsigned int u = __float_as_uint(f);
  return (unsigned short)((u + 0x7FFFu + ((u >> 16) & 1u)) >> 16);
}

// Fast sigmoid: v_exp + v_rcp (error far below bf16-matmul quantization)
__device__ inline float fast_sigmoid(float x) {
#if defined(__HIP_DEVICE_COMPILE__)
  return __builtin_amdgcn_rcpf(1.0f + __expf(-x));
#else
  return 1.0f / (1.0f + __expf(-x));
#endif
}

// 16B global -> LDS, async (ASYNCcnt) when available
__device__ inline void copy16_async(unsigned short* l, const unsigned short* g) {
#if HAVE_ASYNC
  typedef int v4i __attribute__((vector_size(16)));
  typedef __attribute__((address_space(1))) v4i* gptr_t;  // pointer to AS(1) v4i
  typedef __attribute__((address_space(3))) v4i* lptr_t;  // pointer to AS(3) v4i
  gptr_t gp = (gptr_t)(unsigned long long)(uintptr_t)g;
  lptr_t lp = (lptr_t)(unsigned int)(uintptr_t)l;
  __builtin_amdgcn_global_load_async_to_lds_b128(gp, lp, 0, 0);
#else
  *(uint4*)l = *(const uint4*)g;
#endif
}

__device__ inline void wait_async_all() {
#if HAVE_ASYNC
#if __has_builtin(__builtin_amdgcn_s_wait_asynccnt)
  __builtin_amdgcn_s_wait_asynccnt(0);
#else
  asm volatile("s_wait_asynccnt 0" ::: "memory");
#endif
#endif
}

// Build a 16-elem bf16 fragment from two 16B LDS chunks (ds_load_b128 each)
__device__ inline v16bf mk_frag(const unsigned short* p0, const unsigned short* p1) {
  v8bf lo = *(const v8bf*)p0;
  v8bf hi = *(const v8bf*)p1;
  v16bf r;
#pragma unroll
  for (int e = 0; e < 8; ++e) { r[e] = lo[e]; r[e + 8] = hi[e]; }
  return r;
}

// ---------------------------------------------------------------------------
// Pass 1: copy z1,z2 (f32) into d_out, convert to bf16 into d_ws, emit rk2.
// bf16 copies (16.8MB) then live in the 192MB L2 for the GEMM pass.
// ---------------------------------------------------------------------------
__global__ __launch_bounds__(256) void prep_kernel(const float* __restrict__ z1,
                                                   const float* __restrict__ z2,
                                                   const float* __restrict__ rk,
                                                   float* __restrict__ out,
                                                   unsigned short* __restrict__ ws) {
  long id = (long)blockIdx.x * blockDim.x + threadIdx.x;   // one float4 per thread
  const long g4 = NK / 4;
  const float4* src; float4* dst; unsigned short* w; long i = id;
  if (id < g4) { src = (const float4*)z1; dst = (float4*)(out + ADJ);      w = ws;      }
  else { i -= g4; src = (const float4*)z2; dst = (float4*)(out + ADJ + NK); w = ws + NK; }
  float4 v = src[i];
  dst[i] = v;
  ushort4 h;
  h.x = f2bf(v.x); h.y = f2bf(v.y); h.z = f2bf(v.z); h.w = f2bf(v.w);
  ((ushort4*)w)[i] = h;
  if (id < Kd) {  // rk2 = sigmoid(rk_lgt): tiny, keep full-precision divide
    float r = rk[id];
    out[ADJ + 2 * NK + id] = 1.0f / (1.0f + __expf(-r));
  }
}

// ---------------------------------------------------------------------------
// Pass 2: adj[n,m] = 0.25 * sum_b sigmoid(z1[b,n,:] . z2[b,m,:])
// 128x128 tile / block; 8 waves; wave tile 32x64 (2x4 WMMA 16x16 tiles).
// ---------------------------------------------------------------------------
__global__ __launch_bounds__(256) void gemm_sig_kernel(const unsigned short* __restrict__ zbf,
                                                       float* __restrict__ out) {
  extern __shared__ __align__(16) unsigned short smem[];
  const int tid  = threadIdx.x;
  const int lane = tid & 31;
  const int wv   = tid >> 5;
  const int wM   = wv & 3;          // 32-row strip within tile
  const int wN   = wv >> 2;         // 64-col strip within tile
  const int hlf  = lane >> 4;       // lane half (WMMA operand layout)
  const int l16  = lane & 15;
  const int m0 = blockIdx.y * TILE; // rows from z1
  const int n0 = blockIdx.x * TILE; // cols from z2
  const unsigned short* z1 = zbf;
  const unsigned short* z2 = zbf + NK;

  auto issue = [&](int buf, int b) {  // stage one batch's A/B tiles into LDS
    const unsigned short* g1 = z1 + ((long)b * Nn + m0) * Kd;
    const unsigned short* g2 = z2 + ((long)b * Nn + n0) * Kd;
    unsigned short* lA = smem + buf * BUFE;
    unsigned short* lB = lA + MATE;
#pragma unroll
    for (int i = 0; i < 8; ++i) {
      int idx = tid + i * 256;          // 2048 x 16B chunks per matrix
      int row = idx >> 4;
      int col = (idx & 15) * 8;
      copy16_async(lA + row * LDST + col, g1 + (long)row * Kd + col);
      copy16_async(lB + row * LDST + col, g2 + (long)row * Kd + col);
    }
  };

  const v8f zf = {};
  v8f acc[8];
#pragma unroll
  for (int t = 0; t < 8; ++t) acc[t] = zf;

  issue(0, 0);
  wait_async_all();
  __syncthreads();

  for (int b = 0; b < Bt; ++b) {
    const int buf = b & 1;
    if (b + 1 < Bt) issue(buf ^ 1, b + 1);   // overlap next batch's load with compute

    const unsigned short* A   = smem + buf * BUFE;
    const unsigned short* Bm  = A + MATE;
    const unsigned short* aR0 = A  + (wM * 32 +      l16) * LDST;
    const unsigned short* aR1 = A  + (wM * 32 + 16 + l16) * LDST;
    const unsigned short* bR0 = Bm + (wN * 64 +  0 + l16) * LDST;
    const unsigned short* bR1 = Bm + (wN * 64 + 16 + l16) * LDST;
    const unsigned short* bR2 = Bm + (wN * 64 + 32 + l16) * LDST;
    const unsigned short* bR3 = Bm + (wN * 64 + 48 + l16) * LDST;

    v8f c[8];
#pragma unroll
    for (int t = 0; t < 8; ++t) c[t] = zf;

#pragma unroll
    for (int k0 = 0; k0 < Kd; k0 += 32) {
      // A 16x32 bf16 frag: half 0 -> K[k0..k0+7] & K[k0+16..k0+23]; half 1 -> +8
      const int ao = k0 + hlf * 8;
      v16bf a0 = mk_frag(aR0 + ao, aR0 + ao + 16);
      v16bf a1 = mk_frag(aR1 + ao, aR1 + ao + 16);
      // B 32x16 frag: column n = l16, half 0 -> K[k0..k0+7] & K[k0+16..k0+23]; half 1 -> +8
      const int bo = k0 + hlf * 8;
      v16bf b0 = mk_frag(bR0 + bo, bR0 + bo + 16);
      v16bf b1 = mk_frag(bR1 + bo, bR1 + bo + 16);
      v16bf b2 = mk_frag(bR2 + bo, bR2 + bo + 16);
      v16bf b3 = mk_frag(bR3 + bo, bR3 + bo + 16);
      c[0] = __builtin_amdgcn_wmma_f32_16x16x32_bf16(false, a0, false, b0, (short)0, c[0], false, false);
      c[1] = __builtin_amdgcn_wmma_f32_16x16x32_bf16(false, a0, false, b1, (short)0, c[1], false, false);
      c[2] = __builtin_amdgcn_wmma_f32_16x16x32_bf16(false, a0, false, b2, (short)0, c[2], false, false);
      c[3] = __builtin_amdgcn_wmma_f32_16x16x32_bf16(false, a0, false, b3, (short)0, c[3], false, false);
      c[4] = __builtin_amdgcn_wmma_f32_16x16x32_bf16(false, a1, false, b0, (short)0, c[4], false, false);
      c[5] = __builtin_amdgcn_wmma_f32_16x16x32_bf16(false, a1, false, b1, (short)0, c[5], false, false);
      c[6] = __builtin_amdgcn_wmma_f32_16x16x32_bf16(false, a1, false, b2, (short)0, c[6], false, false);
      c[7] = __builtin_amdgcn_wmma_f32_16x16x32_bf16(false, a1, false, b3, (short)0, c[7], false, false);
    }

    // per-batch sigmoid (v_exp + v_rcp), then accumulate in f32
#pragma unroll
    for (int t = 0; t < 8; ++t)
#pragma unroll
      for (int e = 0; e < 8; ++e)
        acc[t][e] += fast_sigmoid(c[t][e]);

    wait_async_all();
    __syncthreads();
  }

  // C/D layout: VGPR r -> (M = hlf*8 + r, N = l16)
#pragma unroll
  for (int i = 0; i < 2; ++i)
#pragma unroll
    for (int j = 0; j < 4; ++j) {
      v8f s = acc[i * 4 + j];
      const int row = m0 + wM * 32 + i * 16 + hlf * 8;
      const int col = n0 + wN * 64 + j * 16 + l16;
#pragma unroll
      for (int r = 0; r < 8; ++r)
        out[(long)(row + r) * Nn + col] = 0.25f * s[r];
    }
}

// ---------------------------------------------------------------------------
extern "C" void kernel_launch(void* const* d_in, const int* in_sizes, int n_in,
                              void* d_out, int out_size, void* d_ws, size_t ws_size,
                              hipStream_t stream) {
  const float* z1 = (const float*)d_in[0];
  const float* z2 = (const float*)d_in[1];
  const float* rk = (const float*)d_in[2];
  float* out = (float*)d_out;
  unsigned short* ws = (unsigned short*)d_ws;

  (void)hipFuncSetAttribute((const void*)gemm_sig_kernel,
                            hipFuncAttributeMaxDynamicSharedMemorySize, SMEM_BYTES);

  const unsigned prep_blocks = (unsigned)((2 * (NK / 4)) / 256);  // 8192
  prep_kernel<<<prep_blocks, 256, 0, stream>>>(z1, z2, rk, out, ws);

  dim3 grid(Nn / TILE, Nn / TILE);  // 64 x 64 tiles
  gemm_sig_kernel<<<grid, 256, SMEM_BYTES, stream>>>(ws, out);
}
